// MultiHeadSelfAttention_52862457479386
// MI455X (gfx1250) — compile-verified, matrix operand order checked by reading, not run on hardware
//
#include <hip/hip_runtime.h>

// ---------------------------------------------------------------------------
// Multi-head self-attention forward (B=4, S=2048, D=1024, H=16, dk=64) with
// RoPE + causal softmax.  CDNA5 WMMA (v_wmma_f32_16x16x32_f16) + TDM staging.
// ---------------------------------------------------------------------------

typedef __attribute__((ext_vector_type(16))) _Float16 v16h;
typedef __attribute__((ext_vector_type(8)))  _Float16 v8h;
typedef __attribute__((ext_vector_type(8)))  float    v8f;
typedef unsigned int u32x4 __attribute__((ext_vector_type(4)));
typedef int          i32x4 __attribute__((ext_vector_type(4)));
typedef int          i32x8 __attribute__((ext_vector_type(8)));

#define D_MODEL 1024
#define NHEAD   16
#define DK      64
#define SEQ     2048
#define BATCH   4
#define TOK     (BATCH * SEQ)   // 8192 token rows

#define BR 128                  // query rows per attention block (8 waves x 16)
#define BC 32                   // kv tile (WMMA K=32 for the PV gemm)
#define LDP (BC + 8)            // padded LDS row stride (halfs) for Vt / Pst
#define LDK (DK + 8)            // padded LDS row stride (halfs) for Kt

// -ln(10000)/32 : inv_freq(i) = exp(ROPE_C * i), i = pair index within head
#define ROPE_C (-0.2878231366242557f)

#if defined(__has_builtin)
#if __has_builtin(__builtin_amdgcn_tensor_load_to_lds) && \
    __has_builtin(__builtin_amdgcn_s_wait_tensorcnt)
#define HAVE_TDM 1
#endif
#endif

// ---------------------------------------------------------------------------
// Fragment loader for 16-bit WMMA operands with K contiguous in memory.
// A 16x32 layout: lanes 0-15 hold M=lane, K chunks {0..7},{16..23}; lanes
// 16-31 hold M=lane-16, K chunks {8..15},{24..31}.  B (32x16) mirrors this
// with N in place of M, so the same addressing works when `base` walks N.
// ---------------------------------------------------------------------------
__device__ __forceinline__ v16h load_frag(const _Float16* base, int ld, int lane) {
  const _Float16* p = base + (size_t)(lane & 15) * ld + ((lane >> 4) & 1) * 8;
  v8h lo = *(const v8h*)(p);
  v8h hi = *(const v8h*)(p + 16);
  return __builtin_shufflevector(lo, hi, 0, 1, 2, 3, 4, 5, 6, 7,
                                         8, 9, 10, 11, 12, 13, 14, 15);
}

__device__ __forceinline__ v8f wmma_f16(v16h a, v16h b, v8f c) {
  return __builtin_amdgcn_wmma_f32_16x16x32_f16(false, a, false, b,
                                                (short)0, c, false, false);
}

// ---------------------------------------------------------------------------
// f32 -> f16 conversion (x and the four weight matrices)
// ---------------------------------------------------------------------------
__global__ void cvt_f32_to_f16(const float* __restrict__ src,
                               _Float16* __restrict__ dst, int n) {
  int i = blockIdx.x * blockDim.x + threadIdx.x;
  int stride = gridDim.x * blockDim.x;
  for (; i < n; i += stride) dst[i] = (_Float16)src[i];
}

// ---------------------------------------------------------------------------
// Fused QKV projection GEMM: one wave computes a 64x64 output tile (4x4 WMMA
// register tile, 16 wmma per K-step of 32) with RoPE fused in the epilogue
// for Q and K.  Output layout [B,H,S,dk].
// y[t,e] = sum_d x[t,d] * w[e,d]  (w row-major (e,d) == K-contiguous B-frag)
// ---------------------------------------------------------------------------
__global__ __launch_bounds__(32)
void qkv_proj_rope(const _Float16* __restrict__ x16,
                   const _Float16* __restrict__ wq16,
                   const _Float16* __restrict__ wk16,
                   const _Float16* __restrict__ wv16,
                   const int* __restrict__ pos,
                   _Float16* __restrict__ Q,
                   _Float16* __restrict__ K,
                   _Float16* __restrict__ V) {
  const int lane  = threadIdx.x;
  const int mBase = blockIdx.x * 64;   // token rows
  const int nBase = blockIdx.y * 64;   // output-feature columns
  const int which = blockIdx.z;        // 0=Q 1=K 2=V
  const _Float16* w = (which == 0) ? wq16 : (which == 1) ? wk16 : wv16;
  _Float16* out     = (which == 0) ? Q    : (which == 1) ? K    : V;

  v8f acc[4][4] = {};
  for (int k = 0; k < D_MODEL; k += 32) {
    v16h af[4], bf[4];
#pragma unroll
    for (int i = 0; i < 4; ++i)
      af[i] = load_frag(x16 + (size_t)(mBase + 16 * i) * D_MODEL + k, D_MODEL, lane);
#pragma unroll
    for (int j = 0; j < 4; ++j)
      bf[j] = load_frag(w + (size_t)(nBase + 16 * j) * D_MODEL + k, D_MODEL, lane);
#pragma unroll
    for (int i = 0; i < 4; ++i)
#pragma unroll
      for (int j = 0; j < 4; ++j)
        acc[i][j] = wmma_f16(af[i], bf[j], acc[i][j]);
  }

  const int hiHalf = (lane >> 4) & 1;
  const int col    = lane & 15;

#pragma unroll
  for (int i = 0; i < 4; ++i) {
#pragma unroll
    for (int j = 0; j < 4; ++j) {
      const int e  = nBase + j * 16 + col;   // feature index
      const int h  = e >> 6;                 // head
      const int dd = e & 63;                 // dim within head
#pragma unroll
      for (int r = 0; r < 8; ++r) {
        int t  = mBase + i * 16 + r + hiHalf * 8;  // global token row
        int bb = t >> 11;                          // batch
        int s  = t & (SEQ - 1);                    // position
        float val = acc[i][j][r];
        if (which < 2) {                           // RoPE on Q and K
          float partner = __shfl_xor(val, 1, 32);  // even<->odd pair lanes
          float p   = (float)pos[s];
          int   pi  = dd >> 1;
          float ang = p * __expf(ROPE_C * (float)pi);
          float sn, cs;
          __sincosf(ang, &sn, &cs);
          // even: qe*cos - qo*sin ; odd: qe*sin + qo*cos
          val = val * cs + partner * sn * ((dd & 1) ? 1.0f : -1.0f);
        }
        out[(((size_t)bb * NHEAD + h) * SEQ + s) * DK + dd] = (_Float16)val;
      }
    }
  }
}

// ---------------------------------------------------------------------------
// Causal flash attention per (b,h).  Block = 8 waves; wave w owns query rows
// [qBlock + 16w, +16).  K tile staged once per block into LDS via the Tensor
// Data Mover (pad-enabled D# produces the bank-padded layout directly); V is
// staged transposed in LDS so PV B-frags are contiguous ds_load_b128; P goes
// through LDS to re-lay C-layout f32 scores into A-layout f16 fragments.
// ---------------------------------------------------------------------------
__global__ __launch_bounds__(256)
void flash_attn(const _Float16* __restrict__ Q,
                const _Float16* __restrict__ K,
                const _Float16* __restrict__ V,
                _Float16* __restrict__ AO) {
  __shared__ __align__(16) _Float16 Kt[BC][LDK];          // [key][dim] (padded)
  __shared__ __align__(16) _Float16 Vt[DK][LDP];          // [dim][key]
  __shared__ __align__(16) _Float16 Pst[8][16][LDP];      // per-wave P staging

  const int tid    = threadIdx.x;
  const int lane   = tid & 31;
  const int wave   = tid >> 5;
  const int b      = blockIdx.z;
  const int h      = blockIdx.y;
  const int qBlock = blockIdx.x * BR;
  const int qBase  = qBlock + wave * 16;
  const int qLast  = qBase + 15;

  const size_t headOff = (size_t)(b * NHEAD + h) * SEQ * DK;
  const _Float16* Qh = Q + headOff;
  const _Float16* Kh = K + headOff;
  const _Float16* Vh = V + headOff;

  // resident Q fragments: 16 rows x 64 dims -> two A-frags (k 0..31, 32..63)
  v16h qf0 = load_frag(Qh + (size_t)qBase * DK + 0,  DK, lane);
  v16h qf1 = load_frag(Qh + (size_t)qBase * DK + 32, DK, lane);

  v8f o0 = {}, o1 = {}, o2 = {}, o3 = {};
  float m[8], l[8];
#pragma unroll
  for (int r = 0; r < 8; ++r) { m[r] = -3.0e30f; l[r] = 0.0f; }

  const int hiHalf = (lane >> 4) & 1;
  const int col    = lane & 15;
  const int kvEnd  = qBlock + BR;      // causal upper bound for this block

  for (int kv = 0; kv < kvEnd; kv += BC) {
    __syncthreads();                   // previous tile fully consumed

#if defined(HAVE_TDM) && defined(__gfx1250__)
    if (wave == 0) {
      // ---- Tensor DMA: K tile (BC x DK f16, row-major) -> padded LDS ----
      // Group0: count=1 | lds_addr | global_addr | type=2
      unsigned int laddr =
          (unsigned int)(size_t)(__attribute__((address_space(3))) void*)&Kt[0][0];
      unsigned long long ga =
          (unsigned long long)(size_t)(Kh + (size_t)kv * DK);
      u32x4 g0 = { 1u, laddr, (unsigned int)ga,
                   (unsigned int)(ga >> 32) | (2u << 30) };
      // Group1: data_size=2B, pad_enable, pad every 32 DWORDs (one 64-half
      // row) by 4 DWORDs (8 halfs) -> LDS row stride = LDK halfs.
      // tensor_dim0=64, tensor_dim1=SEQ, tile=64x32, dim0_stride=64.
      unsigned int d0 = (1u << 16) | (1u << 20) | (4u << 22) | (3u << 25);
      i32x8 g1 = { (int)d0,
                   (int)(64u << 16),            // tensor_dim0[15:0] @ bits 63:48
                   (int)((unsigned)SEQ << 16),  // tensor_dim1[15:0] @ bits 95:80
                   (int)(64u << 16),            // tile_dim0 @ bits 127:112
                   (int)BC,                     // tile_dim1 @ bits 143:128
                   (int)DK,                     // tensor_dim0_stride[31:0]
                   0, 0 };
      i32x4 z4 = { 0, 0, 0, 0 };
#if __clang_major__ >= 23
      i32x8 z8 = { 0, 0, 0, 0, 0, 0, 0, 0 };
      __builtin_amdgcn_tensor_load_to_lds(g0, g1, z4, z4, z8, 0);
#else
      __builtin_amdgcn_tensor_load_to_lds(g0, g1, z4, z4, 0);
#endif
    }
#else
    // fallback: manual K staging
    for (int idx = tid; idx < BC * DK; idx += 256) {
      int key = idx >> 6;
      int dim = idx & 63;
      Kt[key][dim] = Kh[(size_t)(kv + key) * DK + dim];
    }
#endif

    // cooperatively stage V^T tile: Vt[dim][key]
    for (int idx = tid; idx < BC * DK; idx += 256) {
      int key = idx >> 6;
      int dim = idx & 63;
      Vt[dim][key] = Vh[(size_t)(kv + key) * DK + dim];
    }

#if defined(HAVE_TDM) && defined(__gfx1250__)
    if (wave == 0) __builtin_amdgcn_s_wait_tensorcnt(0);
#endif
    __syncthreads();                   // K (TDM) + V stores visible to all

    if (kv <= qLast) {                 // wave-uniform: skip fully-masked tiles
      // scores S = Q K^T : 16 x 32 = two 16x16 C-tiles, K-dim = dk = 64
      v8f s0 = {}, s1 = {};
      s0 = wmma_f16(qf0, load_frag(&Kt[0][0],   LDK, lane), s0);
      s0 = wmma_f16(qf1, load_frag(&Kt[0][32],  LDK, lane), s0);
      s1 = wmma_f16(qf0, load_frag(&Kt[16][0],  LDK, lane), s1);
      s1 = wmma_f16(qf1, load_frag(&Kt[16][32], LDK, lane), s1);

      const float scale = 0.125f;      // 1/sqrt(64)
      float pr0[8], pr1[8];
#pragma unroll
      for (int r = 0; r < 8; ++r) {
        int qi  = qBase + r + hiHalf * 8;
        int kj0 = kv + col;
        int kj1 = kv + 16 + col;
        float v0 = (kj0 <= qi) ? s0[r] * scale : -3.0e30f;
        float v1 = (kj1 <= qi) ? s1[r] * scale : -3.0e30f;
        float mx = fmaxf(v0, v1);
#pragma unroll
        for (int off = 1; off < 16; off <<= 1)
          mx = fmaxf(mx, __shfl_xor(mx, off, 32));   // row max over 16 lanes
        float mNew    = fmaxf(m[r], mx);
        float rescale = __expf(m[r] - mNew);
        float p0 = (kj0 <= qi) ? __expf(v0 - mNew) : 0.0f;
        float p1 = (kj1 <= qi) ? __expf(v1 - mNew) : 0.0f;
        float rs = p0 + p1;
#pragma unroll
        for (int off = 1; off < 16; off <<= 1)
          rs += __shfl_xor(rs, off, 32);             // row sum over 16 lanes
        l[r] = l[r] * rescale + rs;
        m[r] = mNew;
        o0[r] *= rescale; o1[r] *= rescale; o2[r] *= rescale; o3[r] *= rescale;
        pr0[r] = p0; pr1[r] = p1;
      }

      // re-lay P (C-layout f32) into A-layout f16 via wave-private LDS tile
#pragma unroll
      for (int r = 0; r < 8; ++r) {
        int row = r + hiHalf * 8;
        Pst[wave][row][col]      = (_Float16)pr0[r];
        Pst[wave][row][16 + col] = (_Float16)pr1[r];
      }
      asm volatile("s_wait_dscnt 0" ::: "memory");   // same-wave LDS RAW
      v16h pf = load_frag(&Pst[wave][0][0], LDP, lane);

      // O += P V : B-frags from transposed V tile in LDS
      o0 = wmma_f16(pf, load_frag(&Vt[0][0],  LDP, lane), o0);
      o1 = wmma_f16(pf, load_frag(&Vt[16][0], LDP, lane), o1);
      o2 = wmma_f16(pf, load_frag(&Vt[32][0], LDP, lane), o2);
      o3 = wmma_f16(pf, load_frag(&Vt[48][0], LDP, lane), o3);
    }
  }

  // epilogue: normalize and write attention output as [B,S,D] f16
#pragma unroll
  for (int r = 0; r < 8; ++r) {
    int   qi  = qBase + r + hiHalf * 8;
    float inv = 1.0f / l[r];
    size_t base = ((size_t)b * SEQ + qi) * D_MODEL + h * DK;
    AO[base +  0 + col] = (_Float16)(o0[r] * inv);
    AO[base + 16 + col] = (_Float16)(o1[r] * inv);
    AO[base + 32 + col] = (_Float16)(o2[r] * inv);
    AO[base + 48 + col] = (_Float16)(o3[r] * inv);
  }
}

// ---------------------------------------------------------------------------
// Output projection: out[t,e] = sum_d AO[t,d] * wo[e,d], 64x64 tile per wave,
// f32 result straight to d_out.
// ---------------------------------------------------------------------------
__global__ __launch_bounds__(32)
void out_proj(const _Float16* __restrict__ AO,
              const _Float16* __restrict__ wo16,
              float* __restrict__ out) {
  const int lane  = threadIdx.x;
  const int mBase = blockIdx.x * 64;
  const int nBase = blockIdx.y * 64;

  v8f acc[4][4] = {};
  for (int k = 0; k < D_MODEL; k += 32) {
    v16h af[4], bf[4];
#pragma unroll
    for (int i = 0; i < 4; ++i)
      af[i] = load_frag(AO + (size_t)(mBase + 16 * i) * D_MODEL + k, D_MODEL, lane);
#pragma unroll
    for (int j = 0; j < 4; ++j)
      bf[j] = load_frag(wo16 + (size_t)(nBase + 16 * j) * D_MODEL + k, D_MODEL, lane);
#pragma unroll
    for (int i = 0; i < 4; ++i)
#pragma unroll
      for (int j = 0; j < 4; ++j)
        acc[i][j] = wmma_f16(af[i], bf[j], acc[i][j]);
  }

  const int hiHalf = (lane >> 4) & 1;
  const int col    = lane & 15;
#pragma unroll
  for (int i = 0; i < 4; ++i)
#pragma unroll
    for (int j = 0; j < 4; ++j)
#pragma unroll
      for (int r = 0; r < 8; ++r) {
        int row = mBase + i * 16 + r + hiHalf * 8;
        out[(size_t)row * D_MODEL + nBase + j * 16 + col] = acc[i][j][r];
      }
}

// ---------------------------------------------------------------------------
extern "C" void kernel_launch(void* const* d_in, const int* in_sizes, int n_in,
                              void* d_out, int out_size, void* d_ws, size_t ws_size,
                              hipStream_t stream) {
  const float* x  = (const float*)d_in[0];
  const float* wq = (const float*)d_in[1];
  const float* wk = (const float*)d_in[2];
  const float* wv = (const float*)d_in[3];
  const float* wo = (const float*)d_in[4];
  const int*  pos = (const int*)d_in[5];
  float* out = (float*)d_out;

  char* ws = (char*)d_ws;
  size_t off = 0;
  auto alloc = [&](size_t bytes) {
    char* p = ws + off;
    off += (bytes + 255) & ~(size_t)255;
    return p;
  };
  _Float16* x16  = (_Float16*)alloc((size_t)TOK * D_MODEL * 2);
  _Float16* wq16 = (_Float16*)alloc((size_t)D_MODEL * D_MODEL * 2);
  _Float16* wk16 = (_Float16*)alloc((size_t)D_MODEL * D_MODEL * 2);
  _Float16* wv16 = (_Float16*)alloc((size_t)D_MODEL * D_MODEL * 2);
  _Float16* wo16 = (_Float16*)alloc((size_t)D_MODEL * D_MODEL * 2);
  _Float16* Qh   = (_Float16*)alloc((size_t)TOK * D_MODEL * 2);  // [B,H,S,dk]
  _Float16* Kh   = (_Float16*)alloc((size_t)TOK * D_MODEL * 2);
  _Float16* Vh   = (_Float16*)alloc((size_t)TOK * D_MODEL * 2);
  _Float16* AO   = (_Float16*)alloc((size_t)TOK * D_MODEL * 2);  // [B,S,D]
  (void)ws_size; (void)in_sizes; (void)n_in; (void)out_size;

  // 1) precision conversion
  cvt_f32_to_f16<<<2048, 256, 0, stream>>>(x,  x16,  TOK * D_MODEL);
  cvt_f32_to_f16<<<1024, 256, 0, stream>>>(wq, wq16, D_MODEL * D_MODEL);
  cvt_f32_to_f16<<<1024, 256, 0, stream>>>(wk, wk16, D_MODEL * D_MODEL);
  cvt_f32_to_f16<<<1024, 256, 0, stream>>>(wv, wv16, D_MODEL * D_MODEL);
  cvt_f32_to_f16<<<1024, 256, 0, stream>>>(wo, wo16, D_MODEL * D_MODEL);

  // 2) QKV projection + RoPE (WMMA GEMM, 64x64/wave, fused rotary epilogue)
  {
    dim3 grid(TOK / 64, D_MODEL / 64, 3);
    qkv_proj_rope<<<grid, 32, 0, stream>>>(x16, wq16, wk16, wv16, pos,
                                           Qh, Kh, Vh);
  }

  // 3) causal flash attention (WMMA QK^T and PV, TDM-staged K tiles)
  {
    dim3 grid(SEQ / BR, NHEAD, BATCH);
    flash_attn<<<grid, 256, 0, stream>>>(Qh, Kh, Vh, AO);
  }

  // 4) output projection (WMMA GEMM, 64x64/wave, f32 output)
  {
    dim3 grid(TOK / 64, D_MODEL / 64);
    out_proj<<<grid, 32, 0, stream>>>(AO, wo16, out);
  }
}
